// MatNetCrossMHA_39032662786225
// MI455X (gfx1250) — matrix operand order repaired and checked by reading, not compiled, not run on hardware
//
#include <hip/hip_runtime.h>
#include <math.h>

typedef __attribute__((ext_vector_type(16))) _Float16 v16h;
typedef __attribute__((ext_vector_type(8)))  _Float16 v8h;
typedef __attribute__((ext_vector_type(8)))  float    v8f;
typedef __attribute__((ext_vector_type(4)))  float    v4f;

#define KB 4
#define KM 512
#define KN 512
#define KE 256
#define KH 16
#define KD 16

__device__ __forceinline__ v16h cat8(v8h lo, v8h hi) {
  return __builtin_shufflevector(lo, hi, 0,1,2,3,4,5,6,7,8,9,10,11,12,13,14,15);
}

__device__ __forceinline__ v8h zero8() {
  v8h z;
#pragma unroll
  for (int i = 0; i < 8; ++i) z[i] = (_Float16)0.0f;
  return z;
}

// convert 8 contiguous f32 -> 8 f16
__device__ __forceinline__ v8h cvt8(const float* p) {
  v4f a = *(const v4f*)p;
  v4f b = *(const v4f*)(p + 4);
  v8f f = __builtin_shufflevector(a, b, 0,1,2,3,4,5,6,7);
  return __builtin_convertvector(f, v8h);
}

// A-fragment (16xK row-major f32 source), K-chunk of 32 at kbase.
// lane m = lane&15; elements 0..7 = K kbase+sub.., 8..15 = K kbase+16+sub..
__device__ __forceinline__ v16h fragA_f32(const float* row, int lane, int kbase) {
  int sub = (lane & 16) ? 8 : 0;
  return cat8(cvt8(row + kbase + sub), cvt8(row + kbase + 16 + sub));
}

// A-fragment from f16 row-major source
__device__ __forceinline__ v16h fragA_f16(const _Float16* row, int lane, int kbase) {
  int sub = (lane & 16) ? 8 : 0;
  v8h lo = *(const v8h*)(row + kbase + sub);
  v8h hi = *(const v8h*)(row + kbase + 16 + sub);
  return cat8(lo, hi);
}

// A-fragment with only K=16 real data (upper 16 K zero-padded)
__device__ __forceinline__ v16h fragA_f16_k16(const _Float16* row, int lane) {
  int sub = (lane & 16) ? 8 : 0;
  v8h lo = *(const v8h*)(row + sub);
  return cat8(lo, zero8());
}

// B-fragment (KxN, source row-major over K: elements 0..7 = K kbase+sub.., 8..15 = K kbase+16+sub..)
__device__ __forceinline__ v16h fragB_f32(const float* row, int lane, int kbase) {
  const float* p = row + kbase + ((lane & 16) ? 8 : 0);
  return cat8(cvt8(p), cvt8(p + 16));
}

__device__ __forceinline__ v16h fragB_f16(const _Float16* row, int lane, int kbase) {
  int sub = (lane & 16) ? 8 : 0;
  v8h lo = *(const v8h*)(row + kbase + sub);
  v8h hi = *(const v8h*)(row + kbase + 16 + sub);
  return cat8(lo, hi);
}

// B-fragment with only K=16 real (elements 0..7 = K sub.., elements 8..15 hold K>=16 -> zero)
__device__ __forceinline__ v16h fragB_f16_k16(const _Float16* row, int lane) {
  int sub = (lane & 16) ? 8 : 0;
  v8h lo = *(const v8h*)(row + sub);
  return cat8(lo, zero8());
}

__device__ __forceinline__ v8f wmma_f16(v16h a, v16h b, v8f c) {
  return __builtin_amdgcn_wmma_f32_16x16x32_f16(false, a, false, b, (short)0, c, false, false);
}

// ---------------- Kernel 1: fused Q / KV projections ----------------
// tiles 0..2047   : Q  = q_input @ Wq^T   -> qf16 (B,H,M,D)
// tiles 2048..6143: KV = kv_input @ Wkv^T -> kf16 (B,H,N,D), vT (B,H,D,N)
__global__ void __launch_bounds__(256) k_qkv_proj(
    const float* __restrict__ qin, const float* __restrict__ kvin,
    const float* __restrict__ Wq, const float* __restrict__ Wkv,
    _Float16* __restrict__ qf, _Float16* __restrict__ kf,
    _Float16* __restrict__ vT) {
  const int lane = threadIdx.x & 31;
  const int wv = threadIdx.x >> 5;
  const int t = blockIdx.x * 8 + wv;
  const int col = lane & 15;
  const int r0 = (lane >> 4) * 8;
  v8f acc = {};
  if (t < 2048) {
    const int b = t >> 9;
    const int rem = t & 511;
    const int mt = rem >> 4;
    const int h = rem & 15;
    const float* arow = qin + ((size_t)b * KM + mt * 16 + col) * KE;
    const float* brow = Wq + (size_t)(h * 16 + col) * KE;
#pragma unroll
    for (int kk = 0; kk < 8; ++kk)
      acc = wmma_f16(fragA_f32(arow, lane, kk * 32), fragB_f32(brow, lane, kk * 32), acc);
    _Float16* dst = qf + ((size_t)(b * KH + h) * KM + mt * 16) * KD;
#pragma unroll
    for (int r = 0; r < 8; ++r) dst[(r0 + r) * KD + col] = (_Float16)acc[r];
  } else {
    const int t2 = t - 2048;
    const int b = t2 >> 10;
    const int rem = t2 & 1023;
    const int nt = rem >> 5;
    const int ct = rem & 31;
    const float* arow = kvin + ((size_t)b * KN + nt * 16 + col) * KE;
    const float* brow = Wkv + (size_t)(ct * 16 + col) * KE;
#pragma unroll
    for (int kk = 0; kk < 8; ++kk)
      acc = wmma_f16(fragA_f32(arow, lane, kk * 32), fragB_f32(brow, lane, kk * 32), acc);
    if (ct < 16) {  // K head ct
      _Float16* dst = kf + ((size_t)(b * KH + ct) * KN + nt * 16) * KD;
#pragma unroll
      for (int r = 0; r < 8; ++r) dst[(r0 + r) * KD + col] = (_Float16)acc[r];
    } else {        // V head ct-16, stored transposed (D,N): contiguous store
      const int h = ct - 16;
      v8h pk;
#pragma unroll
      for (int r = 0; r < 8; ++r) pk[r] = (_Float16)acc[r];
      *(v8h*)(vT + ((size_t)(b * KH + h) * KD + col) * KN + nt * 16 + r0) = pk;
    }
  }
}

// ---------------- per-element mix MLP: 2 -> 16(relu) -> 1 ----------------
__device__ __forceinline__ float mlp_mix(float s, float dm,
                                         const float* __restrict__ W1a,
                                         const float* __restrict__ W1b,
                                         const float* __restrict__ B1,
                                         const float* __restrict__ W2, float b2) {
  float acc = b2;
#pragma unroll
  for (int j = 0; j < 16; ++j) {
    float hj = fmaf(s, W1a[j], fmaf(dm, W1b[j], B1[j]));
    hj = fmaxf(hj, 0.0f);
    acc = fmaf(hj, W2[j], acc);
  }
  return acc;
}

// ---------------- Kernel 2: scores + MLP mix + online softmax + probs@V ----
// grid = B*H blocks, 8 waves; each wave owns 4 m-tiles of 16 rows.
__global__ void __launch_bounds__(256) k_attn(
    const _Float16* __restrict__ qf, const _Float16* __restrict__ kf,
    const _Float16* __restrict__ vT, const float* __restrict__ dmat,
    const float* __restrict__ mixW1, const float* __restrict__ mixb1,
    const float* __restrict__ mixW2, const float* __restrict__ mixb2,
    _Float16* __restrict__ attn) {
  __shared__ __align__(32) _Float16 pbuf[8][16][40];  // probs transpose staging, stride 40 halves
  const int lane = threadIdx.x & 31;
  const int wv = threadIdx.x >> 5;
  const int bb = blockIdx.x >> 4;
  const int h = blockIdx.x & 15;
  const int col = lane & 15;
  const int r0 = (lane >> 4) * 8;

  // per-head MLP weights (uniform addresses -> scalar loads)
  float W1a[16], W1b[16], B1[16], W2r[16];
  const float* w1 = mixW1 + h * 32;
#pragma unroll
  for (int j = 0; j < 16; ++j) {
    W1a[j] = w1[j];
    W1b[j] = w1[16 + j];
    B1[j] = mixb1[h * 16 + j];
    W2r[j] = mixW2[h * 16 + j];
  }
  const float b2 = mixb2[h];

  const _Float16* kbaseptr = kf + (size_t)(bb * KH + h) * KN * KD;
  const _Float16* vrow = vT + ((size_t)(bb * KH + h) * KD + col) * KN;

  for (int it = 0; it < 4; ++it) {
    const int mt = wv * 4 + it;
    const int mbase = mt * 16;
    const _Float16* qrow = qf + ((size_t)(bb * KH + h) * KM + mbase + col) * KD;
    const v16h qfrag = fragA_f16_k16(qrow, lane);
    const float* dbase = dmat + ((size_t)bb * KM + mbase + r0) * KN + col;

    float run_m[8], run_s[8];
#pragma unroll
    for (int r = 0; r < 8; ++r) { run_m[r] = -1e30f; run_s[r] = 0.0f; }
    v8f acc = {};

    for (int c = 0; c < 16; ++c) {
      const int n0 = c * 32;
      const _Float16* krow0 = kbaseptr + (size_t)(n0 + col) * KD;
      const v16h kf0 = fragB_f16_k16(krow0, lane);
      const v16h kf1 = fragB_f16_k16(krow0 + 16 * KD, lane);
      v8f z = {};
      v8f s0 = wmma_f16(qfrag, kf0, z);
      v8f s1 = wmma_f16(qfrag, kf1, z);

      float x0[8], x1[8], corr[8];
#pragma unroll
      for (int r = 0; r < 8; ++r) {
        float dm0 = dbase[(size_t)r * KN + n0];
        float dm1 = dbase[(size_t)r * KN + n0 + 16];
        x0[r] = mlp_mix(s0[r] * 0.25f, dm0, W1a, W1b, B1, W2r, b2);
        x1[r] = mlp_mix(s1[r] * 0.25f, dm1, W1a, W1b, B1, W2r, b2);
      }
      // online softmax stats per row (rows live in 16-lane half-wave groups)
#pragma unroll
      for (int r = 0; r < 8; ++r) {
        float lm = fmaxf(x0[r], x1[r]);
#pragma unroll
        for (int mk = 8; mk >= 1; mk >>= 1) lm = fmaxf(lm, __shfl_xor(lm, mk, 32));
        float nm = fmaxf(run_m[r], lm);
        corr[r] = __expf(run_m[r] - nm);
        run_m[r] = nm;
        float p0 = __expf(x0[r] - nm);
        float p1 = __expf(x1[r] - nm);
        x0[r] = p0; x1[r] = p1;
        float ls = p0 + p1;
#pragma unroll
        for (int mk = 8; mk >= 1; mk >>= 1) ls += __shfl_xor(ls, mk, 32);
        run_s[r] = run_s[r] * corr[r] + ls;
      }
      // stage probs (C-fragment is column-striped; A-fragment needs row-major K)
#pragma unroll
      for (int r = 0; r < 8; ++r) {
        pbuf[wv][r0 + r][col] = (_Float16)x0[r];
        pbuf[wv][r0 + r][col + 16] = (_Float16)x1[r];
      }
      __builtin_amdgcn_wave_barrier();
      const v16h afrag = fragA_f16(&pbuf[wv][col][0], lane, 0);
      const v16h bfrag = fragB_f16(vrow, lane, n0);
#pragma unroll
      for (int r = 0; r < 8; ++r) acc[r] *= corr[r];  // flash rescale
      acc = wmma_f16(afrag, bfrag, acc);
      __builtin_amdgcn_wave_barrier();
    }
    // normalize and store to attn (B, M, E=H*D) as f16
    _Float16* dst = attn + ((size_t)bb * KM + mbase + r0) * KE + h * 16 + col;
#pragma unroll
    for (int r = 0; r < 8; ++r) dst[(size_t)r * KE] = (_Float16)(acc[r] / run_s[r]);
  }
}

// ---------------- Kernel 3: output projection attn @ Wout^T -> f32 --------
__global__ void __launch_bounds__(256) k_out_proj(
    const _Float16* __restrict__ attn, const float* __restrict__ Wout,
    float* __restrict__ out) {
  const int lane = threadIdx.x & 31;
  const int wv = threadIdx.x >> 5;
  const int t = blockIdx.x * 8 + wv;   // 2048 tiles: 128 row-tiles x 16 col-tiles
  const int mt = t >> 4;
  const int ct = t & 15;
  const int col = lane & 15;
  const int r0 = (lane >> 4) * 8;
  const _Float16* arow = attn + (size_t)(mt * 16 + col) * KE;
  const float* brow = Wout + (size_t)(ct * 16 + col) * KE;
  v8f acc = {};
#pragma unroll
  for (int kk = 0; kk < 8; ++kk)
    acc = wmma_f16(fragA_f16(arow, lane, kk * 32), fragB_f32(brow, lane, kk * 32), acc);
#pragma unroll
  for (int r = 0; r < 8; ++r)
    out[(size_t)(mt * 16 + r0 + r) * KE + ct * 16 + col] = acc[r];
}

extern "C" void kernel_launch(void* const* d_in, const int* in_sizes, int n_in,
                              void* d_out, int out_size, void* d_ws, size_t ws_size,
                              hipStream_t stream) {
  (void)in_sizes; (void)n_in; (void)out_size; (void)ws_size;
  const float* q_input = (const float*)d_in[0];
  const float* kv_input = (const float*)d_in[1];
  const float* dmat = (const float*)d_in[2];
  const float* Wq = (const float*)d_in[3];
  const float* Wkv = (const float*)d_in[4];
  const float* mixW1 = (const float*)d_in[5];
  const float* mixb1 = (const float*)d_in[6];
  const float* mixW2 = (const float*)d_in[7];
  const float* mixb2 = (const float*)d_in[8];
  const float* Wout = (const float*)d_in[9];
  float* out = (float*)d_out;

  const size_t QLEN = (size_t)KB * KH * KM * KD;  // 524288 halves
  _Float16* qf = (_Float16*)d_ws;
  _Float16* kf = qf + QLEN;
  _Float16* vT = kf + QLEN;
  _Float16* attn = vT + QLEN;

  k_qkv_proj<<<768, 256, 0, stream>>>(q_input, kv_input, Wq, Wkv, qf, kf, vT);
  k_attn<<<KB * KH, 256, 0, stream>>>(qf, kf, vT, dmat, mixW1, mixb1, mixW2, mixb2, attn);
  k_out_proj<<<256, 256, 0, stream>>>(attn, Wout, out);
}
